// DPSNR_25194278158359
// MI455X (gfx1250) — matrix operand order repaired and checked by reading, not compiled, not run on hardware
//
#include <hip/hip_runtime.h>
#include <hip/hip_bf16.h>

// ---------------- problem dims ----------------
#define VOCAB 32000
#define DMODEL 512
#define NPOOL 250000
#define WWIN 64
#define BB 4
#define TT 1024
#define ROWS (BB * TT)            // 4096
#define LOOPS 4
#define LOGITS_ELEMS ((size_t)ROWS * VOCAB)   // 131072000

typedef __attribute__((ext_vector_type(16))) __bf16          v16bf;
typedef __attribute__((ext_vector_type(16))) unsigned short  v16u;
typedef __attribute__((ext_vector_type(8)))  unsigned short  v8u;
typedef __attribute__((ext_vector_type(8)))  float           v8f;
typedef __attribute__((ext_vector_type(4)))  int             v4i;

// gfx1250 async global->LDS path (guarded; falls back to register copies)
// Builtin signature (from hipcc diagnostic): (v4i* global, v4i* lds, imm, imm)
#if defined(__gfx1250__) && __has_builtin(__builtin_amdgcn_global_load_async_to_lds_b128)
#define HAVE_ASYNC_LDS 1
#define AS1P(p) ((__attribute__((address_space(1))) v4i*)(p))
#define AS3P(p) ((__attribute__((address_space(3))) v4i*)(p))
__device__ __forceinline__ void wait_async0() {
#if __has_builtin(__builtin_amdgcn_s_wait_asynccnt)
    __builtin_amdgcn_s_wait_asynccnt(0);
#else
    asm volatile("s_wait_asynccnt 0x0" ::: "memory");
#endif
}
#else
#define HAVE_ASYNC_LDS 0
#endif

// ---------------- small helpers ----------------
__device__ __forceinline__ unsigned short f2bf(float f) {
    unsigned int u = __float_as_uint(f);
    return (unsigned short)((u + 0x7FFFu + ((u >> 16) & 1u)) >> 16);
}
__device__ __forceinline__ float gelu_t(float x) {
    float x3 = x * x * x;
    return 0.5f * x * (1.0f + tanhf(0.7978845608028654f * (x + 0.044715f * x3)));
}
__device__ __forceinline__ float sigm(float x) { return 1.0f / (1.0f + expf(-x)); }

// ---------------- generic bf16 WMMA GEMM ----------------
// C[M,N] = A[M,K](bf16,row-major) @ B[K,N](bf16,row-major) + bias[N]
// mode 0: write f32 to outF ; mode 1: gelu -> bf16 to outB
// Block tile 128x128, 8 waves (4x2), each wave 32x64 = 2x4 wmma 16x16 tiles, K step 32.
// A tile staged via async global->LDS (ASYNCcnt); B tile transposed via register->ds.
__global__ __launch_bounds__(256) void wmma_gemm_bf16(
    const unsigned short* __restrict__ A, const unsigned short* __restrict__ Bm,
    const float* __restrict__ bias, float* __restrict__ outF,
    unsigned short* __restrict__ outB, int M, int N, int K, int mode) {
    __shared__ __align__(32) unsigned short ldsA[128 * 32];  // row-major [m][k]
    __shared__ __align__(32) unsigned short ldsB[128 * 32];  // transposed: [n][k]

    const int tid = threadIdx.x;
    const int wave = tid >> 5, lane = tid & 31;
    const int half = lane >> 4, ln15 = lane & 15;
    const int wave_m = wave >> 1, wave_n = wave & 1;
    const int blk_m = blockIdx.y, blk_n = blockIdx.x;

    const v8f zero = {0.f, 0.f, 0.f, 0.f, 0.f, 0.f, 0.f, 0.f};
    v8f acc[2][4];
#pragma unroll
    for (int tm = 0; tm < 2; tm++)
#pragma unroll
        for (int tn = 0; tn < 4; tn++) acc[tm][tn] = zero;

    const int krow = tid >> 3;          // 0..31  (B staging)
    const int nseg = (tid & 7) << 4;    // 0..112
    const int arow = tid >> 1;          // 0..127 (A staging)
    const int aseg = (tid & 1) << 4;    // 0 or 16 halves

    for (int k0 = 0; k0 < K; k0 += 32) {
        __syncthreads();
        // ---- stage A tile [blk_m*128:+128, k0:k0+32] into LDS row-major
        {
            const unsigned short* gap =
                A + (size_t)(blk_m * 128 + arow) * K + k0 + aseg;
            unsigned short* lap = &ldsA[arow * 32 + aseg];
#if HAVE_ASYNC_LDS
            __builtin_amdgcn_global_load_async_to_lds_b128(AS1P(gap), AS3P(lap), 0, 0);
            __builtin_amdgcn_global_load_async_to_lds_b128(AS1P(gap + 8), AS3P(lap + 8), 0, 0);
#else
            v8u t0 = *(const v8u*)gap;
            v8u t1 = *(const v8u*)(gap + 8);
            *(v8u*)lap = t0;
            *(v8u*)(lap + 8) = t1;
#endif
        }
        // ---- stage B tile [k0:k0+32, blk_n*128:+128] into LDS transposed [n][k]
        {
            const unsigned short* bp =
                Bm + (size_t)(k0 + krow) * N + blk_n * 128 + nseg;
            v8u b0 = *(const v8u*)bp;
            v8u b1 = *(const v8u*)(bp + 8);
#pragma unroll
            for (int h = 0; h < 8; h++) {
                ldsB[(nseg + h) * 32 + krow]     = b0[h];
                ldsB[(nseg + 8 + h) * 32 + krow] = b1[h];
            }
            if (k0 + 32 < K)  // hint next B tile (global_prefetch_b8)
                __builtin_prefetch(bp + (size_t)32 * N, 0, 1);
        }
#if HAVE_ASYNC_LDS
        wait_async0();   // drain this wave's ASYNCcnt before the barrier
#endif
        __syncthreads();

        // ---- A fragments from LDS lane = M row, 16 contiguous K per half-wave
        v16bf afr[2];
#pragma unroll
        for (int tm = 0; tm < 2; tm++) {
            const int rloc = wave_m * 32 + tm * 16 + ln15;
            const unsigned short* ap = &ldsA[rloc * 32 + 16 * half];
            v8u c0 = *(const v8u*)ap;          // K = 16*half + 0..7
            v8u c1 = *(const v8u*)(ap + 8);    // K = 16*half + 8..15
            v16u au;
#pragma unroll
            for (int i = 0; i < 8; i++) { au[i] = c0[i]; au[8 + i] = c1[i]; }
            afr[tm] = __builtin_bit_cast(v16bf, au);
        }

        // ---- B fragments from LDS: lane = N column, 16 contiguous K per half-wave
        v16bf bfr[4];
#pragma unroll
        for (int tn = 0; tn < 4; tn++) {
            const int nloc = wave_n * 64 + tn * 16 + ln15;
            v16u bu = *(const v16u*)&ldsB[nloc * 32 + 16 * half];
            bfr[tn] = __builtin_bit_cast(v16bf, bu);
        }

#pragma unroll
        for (int tm = 0; tm < 2; tm++)
#pragma unroll
            for (int tn = 0; tn < 4; tn++)
                acc[tm][tn] = __builtin_amdgcn_wmma_f32_16x16x32_bf16(
                    false, afr[tm], false, bfr[tn], (short)0, acc[tm][tn],
                    false, false);
    }

    // ---- epilogue (C layout: VGPR v -> M = v + 8*half, N = lane&15)
#pragma unroll
    for (int tm = 0; tm < 2; tm++)
#pragma unroll
        for (int tn = 0; tn < 4; tn++) {
            const int rbase = blk_m * 128 + wave_m * 32 + tm * 16 + 8 * half;
            const int col = blk_n * 128 + wave_n * 64 + tn * 16 + ln15;
            const float bv = bias[col];
#pragma unroll
            for (int v = 0; v < 8; v++) {
                float x = acc[tm][tn][v] + bv;
                const size_t off = (size_t)(rbase + v) * N + col;
                if (mode == 0) outF[off] = x;
                else           outB[off] = f2bf(gelu_t(x));
            }
        }
}

// ---------------- f32 -> bf16 converter ----------------
__global__ __launch_bounds__(256) void cvt_bf16(const float* __restrict__ in,
                                                unsigned short* __restrict__ out,
                                                int n) {
    int i = blockIdx.x * 256 + threadIdx.x;
    if (i < n) out[i] = f2bf(in[i]);
}

// ---------------- embedding gather -> bf16 ----------------
__global__ __launch_bounds__(256) void gather_embed(
    const int* __restrict__ ids, const float* __restrict__ embed,
    unsigned short* __restrict__ X) {
    int i = blockIdx.x * 256 + threadIdx.x;           // ROWS*DMODEL
    int row = i >> 9, d = i & 511;
    int id = ids[row];
    X[i] = f2bf(embed[(size_t)id * DMODEL + d]);
}

// ---------------- gelu + LayerNorm over rows (encoder epilogue) ------------
__global__ __launch_bounds__(256) void gelu_ln_rows(
    const float* __restrict__ pre, const float* __restrict__ s,
    const float* __restrict__ b, float* __restrict__ state,
    unsigned short* __restrict__ statebf) {
    __shared__ float r1[256], r2[256];
    const int row = blockIdx.x, tid = threadIdx.x;
    const size_t base = (size_t)row * DMODEL;
    float g0 = gelu_t(pre[base + tid]);
    float g1 = gelu_t(pre[base + 256 + tid]);
    r1[tid] = g0 + g1;
    r2[tid] = g0 * g0 + g1 * g1;
    __syncthreads();
    for (int st = 128; st; st >>= 1) {
        if (tid < st) { r1[tid] += r1[tid + st]; r2[tid] += r2[tid + st]; }
        __syncthreads();
    }
    const float mean = r1[0] * (1.0f / DMODEL);
    const float var = r2[0] * (1.0f / DMODEL) - mean * mean;
    const float rs = rsqrtf(var + 1e-5f);
    float y0 = (g0 - mean) * rs * s[tid] + b[tid];
    float y1 = (g1 - mean) * rs * s[tid + 256] + b[tid + 256];
    state[base + tid] = y0;        statebf[base + tid] = f2bf(y0);
    state[base + 256 + tid] = y1;  statebf[base + 256 + tid] = f2bf(y1);
}

// ---------------- mean-pool over T ----------------
__global__ __launch_bounds__(256) void pool_kernel(const float* __restrict__ state,
                                                   float* __restrict__ pooled) {
    int i = blockIdx.x * 256 + threadIdx.x;  // BB*DMODEL = 2048
    int bb = i >> 9, d = i & 511;
    float s = 0.f;
    for (int t = 0; t < TT; t++) s += state[((size_t)(bb * TT + t)) * DMODEL + d];
    pooled[i] = s * (1.0f / TT);
}

// ---------------- controller: h, mu, sigma, window gaussian retrieve -------
__global__ __launch_bounds__(512) void ctrl_kernel(
    const float* __restrict__ pooled, const float* __restrict__ idx_w,
    const float* __restrict__ idx_b, const float* __restrict__ mu_w,
    const float* __restrict__ mu_b, const float* __restrict__ sig_w,
    const float* __restrict__ sig_b, const float* __restrict__ pool_table,
    float* __restrict__ retr, float* __restrict__ d_out_f, int loop) {
    __shared__ float pl[512], red[512], wts[WWIN], sc[1];
    const int bb = blockIdx.x, tid = threadIdx.x;
    pl[tid] = pooled[bb * DMODEL + tid];
    __syncthreads();

    float a = idx_b[tid];
    for (int k = 0; k < DMODEL; k++) a += pl[k] * idx_w[k * DMODEL + tid];
    const float h = gelu_t(a);

    // mu = sigmoid(h . mu_w + mu_b)
    red[tid] = h * mu_w[tid];
    __syncthreads();
    for (int st = 256; st; st >>= 1) {
        if (tid < st) red[tid] += red[tid + st];
        __syncthreads();
    }
    const float mu = sigm(red[0] + mu_b[0]);
    __syncthreads();

    // sigma = softplus(h . sig_w + sig_b) + 1e-4
    red[tid] = h * sig_w[tid];
    __syncthreads();
    for (int st = 256; st; st >>= 1) {
        if (tid < st) red[tid] += red[tid + st];
        __syncthreads();
    }
    const float sigma = logf(1.0f + expf(red[0] + sig_b[0])) + 1e-4f;

    int start = (int)floorf(mu * (float)NPOOL - (float)(WWIN / 2));
    start = start < 0 ? 0 : (start > NPOOL - WWIN ? NPOOL - WWIN : start);

    if (tid < WWIN) {
        float pos = (float)(start + tid) * (1.0f / NPOOL);
        float z = (pos - mu) / sigma;
        wts[tid] = expf(-0.5f * z * z);
    }
    __syncthreads();
    if (tid == 0) {
        float s = 0.f;
        for (int i = 0; i < WWIN; i++) s += wts[i];
        sc[0] = 1.0f / s;
        // record index in output tail (int bits in the float buffer)
        ((int*)d_out_f)[LOGITS_ELEMS + (size_t)bb * LOOPS + loop] = start;
    }
    __syncthreads();
    const float inv = sc[0];
    float r = 0.f;
    for (int i = 0; i < WWIN; i++)
        r += wts[i] * inv * pool_table[(size_t)(start + i) * DMODEL + tid];
    retr[bb * DMODEL + tid] = r;
}

// ---------------- build combined [state | retrieved] as bf16 ---------------
__global__ __launch_bounds__(256) void build_combined(
    const float* __restrict__ state, const float* __restrict__ retr,
    unsigned short* __restrict__ comb) {
    int i = blockIdx.x * 256 + threadIdx.x;  // ROWS*1024
    int row = i >> 10, c = i & 1023;
    float v;
    if (c < DMODEL) v = state[(size_t)row * DMODEL + c];
    else            v = retr[(row >> 10) * DMODEL + (c - DMODEL)];
    comb[i] = f2bf(v);
}

// ---------------- LN + residual + ACT halting ----------------
__global__ __launch_bounds__(256) void ln_act_rows(
    const float* __restrict__ pre, const float* __restrict__ ln_s,
    const float* __restrict__ ln_b, const float* __restrict__ halt_w,
    const float* __restrict__ halt_b, float* __restrict__ state,
    unsigned short* __restrict__ statebf, float* __restrict__ halt_prob,
    float* __restrict__ halted) {
    __shared__ float r1[256], r2[256];
    const int row = blockIdx.x, tid = threadIdx.x;
    const size_t base = (size_t)row * DMODEL;
    const float p0 = pre[base + tid], p1 = pre[base + 256 + tid];
    const float hp = halt_prob[row], hd = halted[row];

    r1[tid] = p0 + p1;
    r2[tid] = p0 * p0 + p1 * p1;
    __syncthreads();
    for (int st = 128; st; st >>= 1) {
        if (tid < st) { r1[tid] += r1[tid + st]; r2[tid] += r2[tid + st]; }
        __syncthreads();
    }
    const float mean = r1[0] * (1.0f / DMODEL);
    const float var = r2[0] * (1.0f / DMODEL) - mean * mean;
    const float rs = rsqrtf(var + 1e-5f);
    __syncthreads();

    const float prev0 = state[base + tid], prev1 = state[base + 256 + tid];
    const float c0 = prev0 + (p0 - mean) * rs * ln_s[tid] + ln_b[tid];
    const float c1 = prev1 + (p1 - mean) * rs * ln_s[tid + 256] + ln_b[tid + 256];

    // p_halt = sigmoid(candidate . halt_w + halt_b)
    r1[tid] = c0 * halt_w[tid] + c1 * halt_w[tid + 256];
    __syncthreads();
    for (int st = 128; st; st >>= 1) {
        if (tid < st) r1[tid] += r1[tid + st];
        __syncthreads();
    }
    const float p_halt = sigm(r1[0] + halt_b[0]);
    const float hp_new = hp + p_halt * (1.0f - hd);

    const float s0 = (1.0f - hd) * c0 + hd * prev0;
    const float s1 = (1.0f - hd) * c1 + hd * prev1;
    state[base + tid] = s0;        statebf[base + tid] = f2bf(s0);
    state[base + 256 + tid] = s1;  statebf[base + 256 + tid] = f2bf(s1);
    __syncthreads();
    if (tid == 0) {
        halt_prob[row] = hp_new;
        halted[row] = hp_new > 0.99f ? 1.0f : 0.0f;
    }
}

// =====================================================================
extern "C" void kernel_launch(void* const* d_in, const int* in_sizes, int n_in,
                              void* d_out, int out_size, void* d_ws,
                              size_t ws_size, hipStream_t stream) {
    const int*   input_ids = (const int*)d_in[0];
    const float* embed  = (const float*)d_in[1];
    const float* enc_w  = (const float*)d_in[2];
    const float* enc_b  = (const float*)d_in[3];
    const float* ln_es  = (const float*)d_in[4];
    const float* ln_eb  = (const float*)d_in[5];
    const float* idx_w  = (const float*)d_in[6];
    const float* idx_b  = (const float*)d_in[7];
    const float* mu_w   = (const float*)d_in[8];
    const float* mu_b   = (const float*)d_in[9];
    const float* sig_w  = (const float*)d_in[10];
    const float* sig_b  = (const float*)d_in[11];
    const float* pool_t = (const float*)d_in[12];
    const float* int_w1 = (const float*)d_in[13];
    const float* int_b1 = (const float*)d_in[14];
    const float* int_w2 = (const float*)d_in[15];
    const float* int_b2 = (const float*)d_in[16];
    const float* ln_s   = (const float*)d_in[17];
    const float* ln_b   = (const float*)d_in[18];
    const float* halt_w = (const float*)d_in[19];
    const float* halt_b = (const float*)d_in[20];
    const float* dec_w  = (const float*)d_in[21];
    const float* dec_b  = (const float*)d_in[22];
    float* out = (float*)d_out;

    // ---- workspace layout
    char* ws = (char*)d_ws;
    size_t o = 0;
    auto alloc = [&](size_t bytes) {
        size_t r = o;
        o += (bytes + 255) & ~(size_t)255;
        return r;
    };
    const size_t o_encw = alloc((size_t)DMODEL * DMODEL * 2);
    const size_t o_w1   = alloc((size_t)2 * DMODEL * DMODEL * 2);
    const size_t o_w2   = alloc((size_t)DMODEL * DMODEL * 2);
    const size_t o_decw = alloc((size_t)DMODEL * VOCAB * 2);
    const size_t o_X    = alloc((size_t)ROWS * DMODEL * 2);
    const size_t o_st   = alloc((size_t)ROWS * DMODEL * 4);
    const size_t o_stbf = alloc((size_t)ROWS * DMODEL * 2);
    const size_t o_pre  = alloc((size_t)ROWS * DMODEL * 4);
    const size_t o_comb = alloc((size_t)ROWS * 2 * DMODEL * 2);
    const size_t o_int  = alloc((size_t)ROWS * DMODEL * 2);
    const size_t o_pool = alloc((size_t)BB * DMODEL * 4);
    const size_t o_retr = alloc((size_t)BB * DMODEL * 4);
    const size_t o_hp   = alloc((size_t)ROWS * 4);
    const size_t o_hd   = alloc((size_t)ROWS * 4);
    (void)ws_size;

    unsigned short* encw_bf = (unsigned short*)(ws + o_encw);
    unsigned short* w1_bf   = (unsigned short*)(ws + o_w1);
    unsigned short* w2_bf   = (unsigned short*)(ws + o_w2);
    unsigned short* decw_bf = (unsigned short*)(ws + o_decw);
    unsigned short* Xbf     = (unsigned short*)(ws + o_X);
    float*          state   = (float*)(ws + o_st);
    unsigned short* statebf = (unsigned short*)(ws + o_stbf);
    float*          preb    = (float*)(ws + o_pre);
    unsigned short* comb    = (unsigned short*)(ws + o_comb);
    unsigned short* interm  = (unsigned short*)(ws + o_int);
    float*          pooled  = (float*)(ws + o_pool);
    float*          retr    = (float*)(ws + o_retr);
    float*          hp      = (float*)(ws + o_hp);
    float*          hd      = (float*)(ws + o_hd);

    (void)hipMemsetAsync(hp, 0, (size_t)ROWS * 4, stream);
    (void)hipMemsetAsync(hd, 0, (size_t)ROWS * 4, stream);

    // ---- weight conversions (deterministic each call; no caching)
    cvt_bf16<<<(DMODEL * DMODEL + 255) / 256, 256, 0, stream>>>(enc_w, encw_bf, DMODEL * DMODEL);
    cvt_bf16<<<(2 * DMODEL * DMODEL + 255) / 256, 256, 0, stream>>>(int_w1, w1_bf, 2 * DMODEL * DMODEL);
    cvt_bf16<<<(DMODEL * DMODEL + 255) / 256, 256, 0, stream>>>(int_w2, w2_bf, DMODEL * DMODEL);
    cvt_bf16<<<(DMODEL * VOCAB + 255) / 256, 256, 0, stream>>>(dec_w, decw_bf, DMODEL * VOCAB);

    // ---- encode
    gather_embed<<<(ROWS * DMODEL) / 256, 256, 0, stream>>>(input_ids, embed, Xbf);
    wmma_gemm_bf16<<<dim3(DMODEL / 128, ROWS / 128), 256, 0, stream>>>(
        Xbf, encw_bf, enc_b, preb, nullptr, ROWS, DMODEL, DMODEL, 0);
    gelu_ln_rows<<<ROWS, 256, 0, stream>>>(preb, ln_es, ln_eb, state, statebf);

    // ---- ACT loop (fixed, unrolled on host)
    for (int l = 0; l < LOOPS; l++) {
        pool_kernel<<<(BB * DMODEL) / 256, 256, 0, stream>>>(state, pooled);
        ctrl_kernel<<<BB, 512, 0, stream>>>(pooled, idx_w, idx_b, mu_w, mu_b,
                                            sig_w, sig_b, pool_t, retr, out, l);
        build_combined<<<(ROWS * 2 * DMODEL) / 256, 256, 0, stream>>>(state, retr, comb);
        wmma_gemm_bf16<<<dim3(DMODEL / 128, ROWS / 128), 256, 0, stream>>>(
            comb, w1_bf, int_b1, nullptr, interm, ROWS, DMODEL, 2 * DMODEL, 1);
        wmma_gemm_bf16<<<dim3(DMODEL / 128, ROWS / 128), 256, 0, stream>>>(
            interm, w2_bf, int_b2, preb, nullptr, ROWS, DMODEL, DMODEL, 0);
        ln_act_rows<<<ROWS, 256, 0, stream>>>(preb, ln_s, ln_b, halt_w, halt_b,
                                              state, statebf, hp, hd);
    }

    // ---- decoder: [4096,512] @ [512,32000] -> logits f32 into d_out
    wmma_gemm_bf16<<<dim3(VOCAB / 128, ROWS / 128), 256, 0, stream>>>(
        statebf, decw_bf, dec_b, out, nullptr, ROWS, VOCAB, DMODEL, 0);
}
